// PortfolioCrossformer_5592047419478
// MI455X (gfx1250) — matrix operand order repaired and checked by reading, not compiled, not run on hardware
//
#include <hip/hip_runtime.h>
#include <math.h>

// ---------------------------------------------------------------------------
// Types for CDNA5 WMMA (wave32, 16x16x32 f16 -> f32 accumulate)
// ---------------------------------------------------------------------------
typedef _Float16 v16h __attribute__((ext_vector_type(16)));
typedef _Float16 v8h  __attribute__((ext_vector_type(8)));
typedef _Float16 v4h  __attribute__((ext_vector_type(4)));
typedef float    v8f  __attribute__((ext_vector_type(8)));
typedef float    v4f  __attribute__((ext_vector_type(4)));

#define GEMM_BM 128
#define GEMM_BN 128
#define GEMM_BK 32
#define LDS_PAD 8           // 8 x f16 = 16 bytes -> row stride 80B (16B aligned, conflict-spread)
#define ACT_NONE 0
#define ACT_GELU 1

#define NHEAD 8
#define HDIM 64
#define DM 512
#define DFF 1024
#define FAC 10

// Branchless GELU (tanh form): no exec-mask churn, single v_exp_f32.
__device__ __forceinline__ float gelu_f(float v)
{
    float u = v * (0.7978845608028654f + 0.035677408136300125f * v * v);
    float e = __expf(2.0f * u);
    float th = 1.0f - 2.0f / (e + 1.0f);
    return 0.5f * v * (1.0f + th);
}

// ---------------------------------------------------------------------------
// Weight pre-pass: W[K,N] f32 -> Bt[Npad,Kpad] f16, transposed + zero padded.
// grid = (ceil(Kpad/256), Npad), block = 256. No division anywhere.
// ---------------------------------------------------------------------------
__global__ void convert_weight_kernel(const float* __restrict__ W, _Float16* __restrict__ Bt,
                                      int K, int N, int Kpad)
{
    int k = blockIdx.x * 256 + threadIdx.x;
    int n = blockIdx.y;
    if (k >= Kpad) return;
    float v = (k < K && n < N) ? W[(size_t)k * N + n] : 0.0f;
    Bt[(size_t)n * Kpad + k] = (_Float16)v;
}

// ---------------------------------------------------------------------------
// GEMM: C[M,N] = act(A[M,K] @ B + bias), A f32, B pre-staged f16 transposed.
// 256 threads = 8 waves (4x2); each wave owns 32x64 (2x4 WMMA tiles) of the
// 128x128 block tile. Software-pipelined: next tile's global loads are issued
// while the current tile's WMMAs run (waits land before the ds_stores).
// ---------------------------------------------------------------------------
__global__ __launch_bounds__(256) void wmma_gemm_kernel(
    const float* __restrict__ A, const _Float16* __restrict__ Bt,
    const float* __restrict__ bias, float* __restrict__ C,
    int M, int N, int K, int Kpad, int act)
{
    __shared__ __attribute__((aligned(16))) _Float16 As[GEMM_BM][GEMM_BK + LDS_PAD];
    __shared__ __attribute__((aligned(16))) _Float16 Bs[GEMM_BN][GEMM_BK + LDS_PAD]; // Bs[n][k]

    __builtin_assume(Kpad >= GEMM_BK);
    __builtin_assume((Kpad & (GEMM_BK - 1)) == 0);

    const int tid  = threadIdx.x;
    const int wave = tid >> 5;
    const int lane = tid & 31;
    const int hlf  = lane >> 4;        // 16-lane half
    const int lrow = lane & 15;
    const int wm   = (wave >> 1) * 32; // wave tile origin inside 128x128 block tile
    const int wn   = (wave & 1) * 64;

    const int bm0 = blockIdx.y * GEMM_BM;
    const int bn0 = blockIdx.x * GEMM_BN;

    // staging registers (prefetched tile)
    v4f aReg[4];
    v8h bReg[2];

    auto load_tile = [&](int k0) {
        #pragma unroll
        for (int i = 0; i < 4; ++i) {
            int idx = tid + i * 256;            // 1024 float4 slots of 128x32 A tile
            int r  = idx >> 3;
            int c4 = idx & 7;
            int gr = bm0 + r; if (gr > M - 1) gr = M - 1;   // clamp: stays in-bounds
            aReg[i] = *(const v4f*)(A + (size_t)gr * K + k0 + c4 * 4);
        }
        #pragma unroll
        for (int i = 0; i < 2; ++i) {
            int idx = tid + i * 256;            // 512 v8h slots of 128x32 B tile
            int n = idx >> 2;
            int g = idx & 3;
            bReg[i] = *(const v8h*)(Bt + (size_t)(bn0 + n) * Kpad + k0 + g * 8);
        }
    };
    auto store_tile = [&]() {
        #pragma unroll
        for (int i = 0; i < 4; ++i) {
            int idx = tid + i * 256;
            int r  = idx >> 3;
            int c4 = idx & 7;
            v4h h;
            h[0] = (_Float16)aReg[i][0]; h[1] = (_Float16)aReg[i][1];
            h[2] = (_Float16)aReg[i][2]; h[3] = (_Float16)aReg[i][3];
            *(v4h*)&As[r][c4 * 4] = h;
        }
        #pragma unroll
        for (int i = 0; i < 2; ++i) {
            int idx = tid + i * 256;
            int n = idx >> 2;
            int g = idx & 3;
            *(v8h*)&Bs[n][g * 8] = bReg[i];
        }
    };

    v8f acc[2][4] = {};
    load_tile(0);

    for (int k0 = 0; k0 < Kpad; k0 += GEMM_BK) {
        store_tile();
        __syncthreads();

        const int kn = k0 + GEMM_BK;
        if (kn < Kpad) load_tile(kn);           // in flight during WMMAs below
        if (kn + GEMM_BK < Kpad) {              // L2 prefetch two tiles ahead
            int r = tid >> 3, c4 = tid & 7;
            int gr = bm0 + r; if (gr > M - 1) gr = M - 1;
            __builtin_prefetch(A + (size_t)gr * K + kn + GEMM_BK + c4 * 4, 0, 1);
        }

        // ---- fragments (ISA 16-bit layouts) ----
        // A 16x32: lanes 0-15 hold K0..7 & K16..23; lanes 16-31 hold K8..15 & K24..31
        // B 32x16: same K split as A (lanes 0-15 K0..7 & K16..23; lanes 16-31 K8..15 & K24..31)
        v16h afrag[2], bfrag[4];
        #pragma unroll
        for (int t = 0; t < 2; ++t) {
            const _Float16* pa = &As[wm + t * 16 + lrow][0];
            v8h alo = *(const v8h*)(pa + hlf * 8);
            v8h ahi = *(const v8h*)(pa + 16 + hlf * 8);
            afrag[t] = __builtin_shufflevector(alo, ahi, 0, 1, 2, 3, 4, 5, 6, 7,
                                               8, 9, 10, 11, 12, 13, 14, 15);
        }
        #pragma unroll
        for (int t = 0; t < 4; ++t) {
            const _Float16* pb = &Bs[wn + t * 16 + lrow][0];
            v8h blo = *(const v8h*)(pb + hlf * 8);
            v8h bhi = *(const v8h*)(pb + 16 + hlf * 8);
            bfrag[t] = __builtin_shufflevector(blo, bhi, 0, 1, 2, 3, 4, 5, 6, 7,
                                               8, 9, 10, 11, 12, 13, 14, 15);
        }

        #pragma unroll
        for (int i = 0; i < 2; ++i)
            #pragma unroll
            for (int j = 0; j < 4; ++j)
                acc[i][j] = __builtin_amdgcn_wmma_f32_16x16x32_f16(
                    false, afrag[i], false, bfrag[j], (short)0, acc[i][j], false, false);
        __syncthreads();
    }

    // ---- epilogue: C layout VGPR r -> M = r + 8*half, N = lane%16 ----
    // bias hoisted: one predicated load per column tile (4 per wave)
    float bv[4];
    int   cols[4];
    #pragma unroll
    for (int j = 0; j < 4; ++j) {
        cols[j] = bn0 + wn + j * 16 + lrow;
        bv[j] = (bias && cols[j] < N) ? bias[cols[j]] : 0.0f;
    }
    #pragma unroll
    for (int i = 0; i < 2; ++i) {
        #pragma unroll
        for (int j = 0; j < 4; ++j) {
            #pragma unroll
            for (int r = 0; r < 8; ++r) {
                int row = bm0 + wm + i * 16 + hlf * 8 + r;
                float v = acc[i][j][r] + bv[j];
                if (act == ACT_GELU) v = gelu_f(v);
                if (row < M && cols[j] < N)
                    C[(size_t)row * N + cols[j]] = v;
            }
        }
    }
}

// ---------------------------------------------------------------------------
// LayerNorm over last dim (512 or 1024), optional residual: y = LN(in + res)
// ---------------------------------------------------------------------------
__global__ __launch_bounds__(256) void layernorm_kernel(
    float* __restrict__ out, const float* __restrict__ in, const float* __restrict__ res,
    const float* __restrict__ gamma, const float* __restrict__ beta, int dim)
{
    const int row = blockIdx.x;
    const int tid = threadIdx.x;
    const float* xr = in + (size_t)row * dim;
    const float* rr = res ? res + (size_t)row * dim : nullptr;

    float vals[4];
    float s = 0.0f, ss = 0.0f;
    int i = 0;
    for (int c = tid; c < dim; c += 256, ++i) {
        float v = xr[c] + (rr ? rr[c] : 0.0f);
        vals[i] = v; s += v; ss += v * v;
    }
    __shared__ float rs[256], rss[256];
    rs[tid] = s; rss[tid] = ss;
    __syncthreads();
    for (int off = 128; off > 0; off >>= 1) {
        if (tid < off) { rs[tid] += rs[tid + off]; rss[tid] += rss[tid + off]; }
        __syncthreads();
    }
    float mean = rs[0] / (float)dim;
    float var  = rss[0] / (float)dim - mean * mean;
    float inv  = rsqrtf(var + 1e-5f);
    i = 0;
    for (int c = tid; c < dim; c += 256, ++i)
        out[(size_t)row * dim + c] = gamma[c] * (vals[i] - mean) * inv + beta[c];
}

// ---------------------------------------------------------------------------
// Attention core: O = softmax(Q K^T / sqrt(64)) V  per (batch, head, query row)
// Q: [bt, Lq, 512], K/V: [bt, S, 512], head h occupies cols h*64..h*64+63
// grid = (Lq, NHEAD, bt), block = 64
// ---------------------------------------------------------------------------
__global__ __launch_bounds__(64) void attn_core_kernel(
    const float* __restrict__ Q, const float* __restrict__ Km, const float* __restrict__ V,
    float* __restrict__ O, int Lq, int S)
{
    const int lq = blockIdx.x, h = blockIdx.y, bt = blockIdx.z;
    const int tid = threadIdx.x;
    __shared__ __attribute__((aligned(16))) float qs[HDIM];
    __shared__ float sc[400];
    __shared__ float red[64];

    const size_t qoff = ((size_t)bt * Lq + lq) * DM + h * HDIM;
    qs[tid] = Q[qoff + tid];
    __syncthreads();

    float lmax = -1e30f;
    for (int j = tid; j < S; j += 64) {
        const v4f* kp4 = (const v4f*)(Km + ((size_t)bt * S + j) * DM + h * HDIM);
        const v4f* qs4 = (const v4f*)qs;
        float d = 0.0f;
        #pragma unroll
        for (int e = 0; e < HDIM / 4; ++e) {
            v4f kk = kp4[e], qq = qs4[e];
            d += qq[0] * kk[0] + qq[1] * kk[1] + qq[2] * kk[2] + qq[3] * kk[3];
        }
        d *= 0.125f;  // 1/sqrt(64)
        sc[j] = d;
        lmax = fmaxf(lmax, d);
    }
    red[tid] = lmax; __syncthreads();
    for (int off = 32; off > 0; off >>= 1) {
        if (tid < off) red[tid] = fmaxf(red[tid], red[tid + off]);
        __syncthreads();
    }
    float mx = red[0];
    __syncthreads();
    float lsum = 0.0f;
    for (int j = tid; j < S; j += 64) { float p = __expf(sc[j] - mx); sc[j] = p; lsum += p; }
    red[tid] = lsum; __syncthreads();
    for (int off = 32; off > 0; off >>= 1) {
        if (tid < off) red[tid] += red[tid + off];
        __syncthreads();
    }
    float inv = 1.0f / red[0];
    __syncthreads();

    float o = 0.0f;
    for (int j = 0; j < S; ++j)
        o += sc[j] * V[((size_t)bt * S + j) * DM + h * HDIM + tid];
    O[qoff + tid] = o * inv;
}

// ---------------------------------------------------------------------------
// Layout plumbing kernels — multi-dim grids, zero divisions
// ---------------------------------------------------------------------------
// x[32,128,50,8] -> A[(b*400+d)*8+s][16] ; grid = (400, 32), block = 128
__global__ void rearrange_embed_kernel(const float* __restrict__ x, float* __restrict__ A)
{
    const int d = blockIdx.x, b = blockIdx.y;
    const int s  = threadIdx.x >> 4;
    const int tt = threadIdx.x & 15;
    const int t  = s * 16 + tt;
    const int n = d >> 3, f = d & 7;
    A[(((size_t)b * 400 + d) * 8 + s) * 16 + tt] =
        x[(((size_t)b * 128 + t) * 50 + n) * 8 + f];
}

// e[b][ds][c] += pos[ds][c] ; grid = (3200, 32), block = 256 (2 ch each)
__global__ void add_pos_kernel(float* __restrict__ e, const float* __restrict__ pos)
{
    const int ds = blockIdx.x, b = blockIdx.y;
    #pragma unroll
    for (int q = 0; q < 2; ++q) {
        int c = threadIdx.x + q * 256;
        e[((size_t)b * 3200 + ds) * DM + c] += pos[(size_t)ds * DM + c];
    }
}

// out[b][y][x][c] = in[b][x][y][c] ; grid = (Y, X, B), block = 256 (2 ch each)
__global__ void permute_kernel(float* __restrict__ out, const float* __restrict__ in,
                               int X, int Y)
{
    const int y = blockIdx.x, x = blockIdx.y, b = blockIdx.z;
    const size_t oi = (((size_t)b * Y + y) * X + x) * DM;
    const size_t ii = (((size_t)b * X + x) * Y + y) * DM;
    #pragma unroll
    for (int q = 0; q < 2; ++q) {
        int c = threadIdx.x + q * 256;
        out[oi + c] = in[ii + c];
    }
}

// out[b][lf][c] = router[lf][c] ; grid = (Lf, B), block = 256 (2 ch each)
__global__ void bcast_router_kernel(float* __restrict__ out, const float* __restrict__ router,
                                    int Lf)
{
    const int lf = blockIdx.x, b = blockIdx.y;
    #pragma unroll
    for (int q = 0; q < 2; ++q) {
        int c = threadIdx.x + q * 256;
        out[((size_t)b * Lf + lf) * DM + c] = router[(size_t)lf * DM + c];
    }
}

// SegMerge gather (win=2): out[bd][l2][c*2+w] = in[bd][2*l2+w][c]
// grid = (4, 400, 32), block = 256 (4 j each)
__global__ void merge_gather_kernel(float* __restrict__ out, const float* __restrict__ in)
{
    const int l2 = blockIdx.x, d = blockIdx.y, b = blockIdx.z;
    const size_t bd = (size_t)b * 400 + d;
    #pragma unroll
    for (int q = 0; q < 4; ++q) {
        int j = threadIdx.x + q * 256;      // 0..1023
        int w = j & 1, c = j >> 1;
        out[(bd * 4 + l2) * 1024 + j] = in[(bd * 8 + 2 * l2 + w) * DM + c];
    }
}

// out[bd][c] = mean_l in[bd][l][c] ; grid = (12800), block = 256 (2 ch each)
__global__ void mean_seg_kernel(float* __restrict__ out, const float* __restrict__ in, int L)
{
    const size_t bd = blockIdx.x;
    const float invL = 1.0f / (float)L;
    #pragma unroll
    for (int q = 0; q < 2; ++q) {
        int c = threadIdx.x + q * 256;
        float s = 0.0f;
        for (int l = 0; l < L; ++l) s += in[(bd * L + l) * DM + c];
        out[bd * DM + c] = s * invL;
    }
}

// ---------------------------------------------------------------------------
// Host-side orchestration
// ---------------------------------------------------------------------------
static _Float16* g_wb = nullptr;   // transient f16 transposed-weight staging buffer

static inline void gemm(const float* A, const float* W, const float* bias, float* C,
                        int M, int N, int K, int act, hipStream_t st)
{
    int Kpad = (K + GEMM_BK - 1) & ~(GEMM_BK - 1);
    int Npad = (N + GEMM_BN - 1) & ~(GEMM_BN - 1);
    convert_weight_kernel<<<dim3((Kpad + 255) / 256, Npad), 256, 0, st>>>(W, g_wb, K, N, Kpad);
    dim3 g(Npad / GEMM_BN, (M + GEMM_BM - 1) / GEMM_BM);
    wmma_gemm_kernel<<<g, 256, 0, st>>>(A, g_wb, bias, C, M, N, K, Kpad, act);
}

static inline void lnorm(float* out, const float* in, const float* res,
                         const float* g, const float* b, int rows, int dim, hipStream_t st)
{
    layernorm_kernel<<<rows, 256, 0, st>>>(out, in, res, g, b, dim);
}

struct AttnP { const float *Wq,*bq,*Wk,*bk,*Wv,*bv,*Wo,*bo; };
struct TsaP {
    const float *m1W1,*m1W2,*m1b1,*m1b2, *m2W1,*m2W2,*m2b1,*m2b2;
    const float *n1g,*n1b,*n2g,*n2b,*n3g,*n3b,*n4g,*n4b;
    AttnP recv; const float* router; AttnP send; AttnP time;
};

static AttnP attn_from(void* const* din, int base)
{
    AttnP p;
    p.Wk = (const float*)din[base + 0]; p.Wo = (const float*)din[base + 1];
    p.Wq = (const float*)din[base + 2]; p.Wv = (const float*)din[base + 3];
    p.bk = (const float*)din[base + 4]; p.bo = (const float*)din[base + 5];
    p.bq = (const float*)din[base + 6]; p.bv = (const float*)din[base + 7];
    return p;
}

static TsaP tsa_from(void* const* din, int base)
{
    TsaP t;
    t.m1W1 = (const float*)din[base + 0]; t.m1W2 = (const float*)din[base + 1];
    t.m1b1 = (const float*)din[base + 2]; t.m1b2 = (const float*)din[base + 3];
    t.m2W1 = (const float*)din[base + 4]; t.m2W2 = (const float*)din[base + 5];
    t.m2b1 = (const float*)din[base + 6]; t.m2b2 = (const float*)din[base + 7];
    t.n1g  = (const float*)din[base + 8];  t.n1b = (const float*)din[base + 9];
    t.n2g  = (const float*)din[base + 10]; t.n2b = (const float*)din[base + 11];
    t.n3g  = (const float*)din[base + 12]; t.n3b = (const float*)din[base + 13];
    t.n4g  = (const float*)din[base + 14]; t.n4b = (const float*)din[base + 15];
    t.recv   = attn_from(din, base + 16);
    t.router = (const float*)din[base + 24];
    t.send   = attn_from(din, base + 25);
    t.time   = attn_from(din, base + 33);
    return t;
}

// full attention layer: out = (softmax(proj_q(qin) proj_k(kvin)^T) proj_v(kvin)) Wo + bo
static void run_attn(const float* qin, int rows_q, const float* kvin, int rows_kv,
                     int bt, int Lq, int S, const AttnP& p,
                     float* qb, float* kb, float* vb, float* ob, float* out, hipStream_t st)
{
    gemm(qin,  p.Wq, p.bq, qb, rows_q,  DM, DM, ACT_NONE, st);
    gemm(kvin, p.Wk, p.bk, kb, rows_kv, DM, DM, ACT_NONE, st);
    gemm(kvin, p.Wv, p.bv, vb, rows_kv, DM, DM, ACT_NONE, st);
    attn_core_kernel<<<dim3(Lq, NHEAD, bt), 64, 0, st>>>(qb, kb, vb, ob, Lq, S);
    gemm(ob, p.Wo, p.bo, out, rows_q, DM, DM, ACT_NONE, st);
}

// TwoStageAttentionLayer, x in/out layout [B, D, L, DM]
static void run_tsa(float* x, int B, int D, int L, const TsaP& t,
                    float* w1, float* w2, float* w3, float* w4, float* hb,
                    float* smr, float* smq, float* smk, float* smv, float* smo, float* smb,
                    hipStream_t st)
{
    const int rows = B * D * L;

    // --- time stage ---
    run_attn(x, rows, x, rows, B * D, L, L, t.time, w1, w2, w3, w4, w1, st);
    lnorm(w2, x, w1, t.n1g, t.n1b, rows, DM, st);
    gemm(w2, t.m1W1, t.m1b1, hb, rows, DFF, DM, ACT_GELU, st);
    gemm(hb, t.m1W2, t.m1b2, w1, rows, DM, DFF, ACT_NONE, st);
    lnorm(w4, w2, w1, t.n2g, t.n2b, rows, DM, st);

    // --- cross-dimension stage ---
    permute_kernel<<<dim3(L, D, B), 256, 0, st>>>(w2, w4, D, L);   // s = [B,L,D,DM]
    bcast_router_kernel<<<dim3(L * FAC, B), 256, 0, st>>>(smr, t.router, L * FAC);
    run_attn(smr, B * L * FAC, w2, rows, B * L, FAC, D, t.send, smq, w1, w3, smo, smb, st);
    run_attn(w2, rows, smb, B * L * FAC, B * L, D, FAC, t.recv, w1, smk, smv, w4, w1, st);
    lnorm(w3, w2, w1, t.n3g, t.n3b, rows, DM, st);
    gemm(w3, t.m2W1, t.m2b1, hb, rows, DFF, DM, ACT_GELU, st);
    gemm(hb, t.m2W2, t.m2b2, w1, rows, DM, DFF, ACT_NONE, st);
    lnorm(w2, w3, w1, t.n4g, t.n4b, rows, DM, st);
    permute_kernel<<<dim3(D, L, B), 256, 0, st>>>(x, w2, L, D);    // back to [B,D,L,DM]
}

extern "C" void kernel_launch(void* const* d_in, const int* in_sizes, int n_in,
                              void* d_out, int out_size, void* d_ws, size_t ws_size,
                              hipStream_t stream)
{
    (void)in_sizes; (void)n_in; (void)out_size; (void)ws_size;

    // ---- parameter pointers (jax pytree order: sorted dict keys, top-level dict order) ----
    const float* x = (const float*)d_in[0];
    TsaP b0 = tsa_from(d_in, 1);                      // params['block0'] leaves 1..41
    const float* mergeW = (const float*)d_in[42];     // block1.merge.W
    const float* mergeb = (const float*)d_in[43];     // block1.merge.b
    const float* mng    = (const float*)d_in[44];     // block1.merge.norm gamma
    const float* mnb    = (const float*)d_in[45];     // block1.merge.norm beta
    TsaP b1 = tsa_from(d_in, 46);                     // block1.tsa leaves 46..86
    const float* embW = (const float*)d_in[87];
    const float* embb = (const float*)d_in[88];
    const float* head = (const float*)d_in[89];
    const float* pos  = (const float*)d_in[90];
    const float* png  = (const float*)d_in[91];
    const float* pnb  = (const float*)d_in[92];

    // ---- workspace layout ----
    float* ws = (float*)d_ws;
    const size_t SZ = (size_t)102400 * DM;            // one big activation region
    float* t0 = ws;
    float* w1 = ws + 1 * SZ;
    float* w2 = ws + 2 * SZ;
    float* w3 = ws + 3 * SZ;
    float* w4 = ws + 4 * SZ;
    float* hb = ws + 5 * SZ;                          // DFF region (2*SZ wide)
    float* sm = ws + 7 * SZ;
    const size_t SMU = (size_t)256 * FAC * DM;        // router-sized sub-regions
    float* smr = sm + 0 * SMU;
    float* smq = sm + 1 * SMU;
    float* smk = sm + 2 * SMU;
    float* smv = sm + 3 * SMU;
    float* smo = sm + 4 * SMU;
    float* smb = sm + 5 * SMU;
    float* ae  = sm + 6 * SMU;                        // embedding A matrix [102400,16]
    // f16 weight staging buffer: sized for head weight (Npad=128, Kpad=204800)
    g_wb = (_Float16*)(ae + (size_t)102400 * 16 + 1024);

    // ---- DSW embedding + pos + pre-norm ----
    rearrange_embed_kernel<<<dim3(400, 32), 128, 0, stream>>>(x, ae);
    gemm(ae, embW, embb, t0, 102400, DM, 16, ACT_NONE, stream);
    add_pos_kernel<<<dim3(3200, 32), 256, 0, stream>>>(t0, pos);
    lnorm(t0, t0, nullptr, png, pnb, 102400, DM, stream);

    // ---- scale block 0 (L = 8) ----
    run_tsa(t0, 32, 400, 8, b0, w1, w2, w3, w4, hb, smr, smq, smk, smv, smo, smb, stream);

    // ---- segment merge (win = 2): [32,400,8,512] -> [32,400,4,512] ----
    merge_gather_kernel<<<dim3(4, 400, 32), 256, 0, stream>>>(w1, t0);
    lnorm(w1, w1, nullptr, mng, mnb, 51200, 1024, stream);
    gemm(w1, mergeW, mergeb, t0, 51200, DM, 1024, ACT_NONE, stream);

    // ---- scale block 1 (L = 4) ----
    run_tsa(t0, 32, 400, 4, b1, w1, w2, w3, w4, hb, smr, smq, smk, smv, smo, smb, stream);

    // ---- mean over segments + head ----
    mean_seg_kernel<<<12800, 256, 0, stream>>>(w1, t0, 4);
    gemm(w1, head, nullptr, (float*)d_out, 32, 50, 204800, ACT_NONE, stream);
}
